// SpatialStructureAttention_17781164605592
// MI455X (gfx1250) — matrix operand order repaired and checked by reading, not compiled, not run on hardware
//
#include <hip/hip_runtime.h>
#include <hip/hip_bf16.h>

// ---------------------------------------------------------------------------
// SpatialStructureAttention for MI455X (gfx1250, wave32, WMMA).
// - All GEMMs on v_wmma_f32_16x16x32_bf16 (fp32 accumulate), 32x32 register
//   block per wave (4 WMMA per K-step with A/B fragment reuse).
// - bf16 LDS staging in the attention kernel uses the Tensor Data Mover
//   (tensor_load_to_lds + s_wait_tensorcnt), with TDM padding reproducing the
//   LDS pitches. bf16 A staging in the output GEMM uses
//   GLOBAL_LOAD_ASYNC_TO_LDS_B128 (ASYNCcnt). fp32 sources stage via VALU
//   convert + global_prefetch.
// ---------------------------------------------------------------------------

typedef __bf16 bf16;
typedef __attribute__((ext_vector_type(16))) __bf16 v16bf;
typedef __attribute__((ext_vector_type(8)))  float  v8f;
typedef __attribute__((ext_vector_type(4)))  unsigned int v4u;
typedef __attribute__((ext_vector_type(8)))  int v8i;
typedef __attribute__((ext_vector_type(4)))  int v4i;

union BFragU {
  v16bf v;
  bf16  h[16];
  uint4 q[2];
};

__device__ __forceinline__ v8f wmma_bf16(v16bf a, v16bf b, v8f c) {
  // (neg_a, A, neg_b, B, c_mod, C, reuse_a, reuse_b)
  return __builtin_amdgcn_wmma_f32_16x16x32_bf16(false, a, false, b, (short)0, c,
                                                 false, false);
}

// ---- Async (per-lane) global->LDS copy, ASYNCcnt tracked --------------------
__device__ __forceinline__ void async_ld_b128(void* lds_dst, const void* gsrc) {
  asm volatile("global_load_async_to_lds_b128 %0, %1, off"
               :
               : "v"((unsigned int)(uintptr_t)lds_dst), "v"(gsrc)
               : "memory");
}
__device__ __forceinline__ void wait_async0() {
  asm volatile("s_wait_asynccnt 0x0" ::: "memory");
}
__device__ __forceinline__ void wait_tensor0() {
  asm volatile("s_wait_tensorcnt 0x0" ::: "memory");
}

// ---- Tensor Data Mover 2D tile load (CDNA5 §8 D# descriptor) ----------------
// Loads tile_d0 x tile_d1 elements (2B each) from a row-major tensor with
// row length stride_elems into LDS at lds_byte_off, inserting pad_amount+1
// DWORDs of padding every 2<<pad_interval DWORDs (reproduces the LDS pitch).
__device__ __forceinline__ void tdm_load_2d(unsigned lds_byte_off,
                                            const void* gaddr,
                                            unsigned tile_d0, unsigned tile_d1,
                                            unsigned tensor_d0, unsigned tensor_d1,
                                            unsigned stride_elems,
                                            unsigned pad_interval,
                                            unsigned pad_amount) {
  unsigned long long ga = (unsigned long long)(uintptr_t)gaddr;
  v4u g0;
  g0[0] = 1u;                                        // count=1, user descriptor
  g0[1] = lds_byte_off;                              // lds_addr (bytes)
  g0[2] = (unsigned)(ga & 0xffffffffu);              // global_addr[31:0]
  g0[3] = (unsigned)((ga >> 32) & 0x01ffffffu) | (2u << 30);  // addr[56:32]|type=2
  v8i g1;
  g1[0] = (int)((1u << 16) |                         // data_size = 2 bytes
                (1u << 20) |                         // pad_enable
                (pad_interval << 22) | (pad_amount << 25));
  g1[1] = (int)((tensor_d0 & 0xffffu) << 16);        // tensor_dim0[15:0]
  g1[2] = (int)(((tensor_d0 >> 16) & 0xffffu) | ((tensor_d1 & 0xffffu) << 16));
  g1[3] = (int)(((tensor_d1 >> 16) & 0xffffu) | ((tile_d0 & 0xffffu) << 16));
  g1[4] = (int)(tile_d1 & 0xffffu);                  // tile_dim1, tile_dim2=0
  g1[5] = (int)stride_elems;                         // tensor_dim0_stride[31:0]
  g1[6] = 0;
  g1[7] = 0;
  v4i z = {0, 0, 0, 0};
#if defined(__clang_major__) && (__clang_major__ >= 23)
  v8i z8 = {0, 0, 0, 0, 0, 0, 0, 0};
  __builtin_amdgcn_tensor_load_to_lds(g0, g1, z, z, z8, 0);
#else
  __builtin_amdgcn_tensor_load_to_lds(g0, g1, z, z, 0);
#endif
}

// A fragment: 16x32 bf16 tile. lanes 0-15 -> M=lane, K base 0; lanes 16-31 ->
// M=lane-16, K base +8. Two 16B-contiguous K runs per lane -> ds_load_b128s.
__device__ __forceinline__ v16bf load_a_frag(const bf16* Abuf, int pitch,
                                             int m_local, int lane) {
  BFragU f;
  const bf16* row = Abuf + m_local * pitch;
  const int koff = (lane & 16) ? 8 : 0;
#pragma unroll
  for (int p = 0; p < 8; ++p) {
    int kk = ((p < 4) ? 0 : 16) + koff + 2 * (p & 3);
    f.h[2 * p]     = row[kk];
    f.h[2 * p + 1] = row[kk + 1];
  }
  return f.v;
}

// B fragment: 32x16 bf16 tile from LDS stored TRANSPOSED, Bt[n][k].
__device__ __forceinline__ v16bf load_b_frag(const bf16* Bt, int pitch,
                                             int n_local, int lane) {
  BFragU f;
  const bf16* base = Bt + n_local * pitch + ((lane & 16) ? 8 : 0);
  f.q[0] = *(const uint4*)(base);
  f.q[1] = *(const uint4*)(base + 16);
  return f.v;
}

// ---------------------------------------------------------------------------
// Generic 512-K, 512-N WMMA GEMM:  out[M,512] = A[M,512] * B[512,512] (+bias)
// WG tile 64x128, 8 waves as 2(m) x 4(n); each wave owns 32x32 (4 accum tiles).
// ---------------------------------------------------------------------------
template <bool AF32, bool BF32, bool BATB, bool BIAS, bool OUTF32>
__global__ __launch_bounds__(256) void gemm512(const void* __restrict__ Ap,
                                               const void* __restrict__ Bp,
                                               const float* __restrict__ bias,
                                               void* __restrict__ outp) {
  constexpr int KDIM = 512, NDIM = 512;
  constexpr int P = 40;  // LDS row pitch in bf16 (80B: keeps 16B alignment)
  __shared__ __align__(16) bf16 Abuf[64 * P];   // 64  x 32 A tile
  __shared__ __align__(16) bf16 Bt[128 * P];    // 128 x 32 B tile (transposed)

  const int m_blk = blockIdx.x * 64;
  const int n_blk = blockIdx.y * 128;
  const int tid  = threadIdx.x;
  const int lane = tid & 31;
  const int w    = tid >> 5;
  const int mw   = w & 1;   // 32-row half
  const int nw   = w >> 1;  // 32-col quarter

  const float* Af = (const float*)Ap;
  const bf16*  Ah = (const bf16*)Ap;
  const float* Bf = (const float*)Bp;
  const bf16*  Bh = (const bf16*)Bp;
  if (BATB) {  // per-batch B (64 rows never straddle a batch: 512 % 64 == 0)
    size_t boff = (size_t)(m_blk >> 9) * KDIM * NDIM;
    Bf += boff;
    Bh += boff;
  }

  v8f acc[2][2] = {};

  for (int k0 = 0; k0 < KDIM; k0 += 32) {
    __syncthreads();
    // ---- stage A tile 64x32 ----
    if constexpr (AF32) {
#pragma unroll
      for (int e = tid; e < 64 * 32; e += 256) {
        int m = e >> 5, kk = e & 31;
        Abuf[m * P + kk] = (bf16)Af[(size_t)(m_blk + m) * KDIM + (k0 + kk)];
      }
      if (k0 + 32 < KDIM)
        __builtin_prefetch(&Af[(size_t)(m_blk + (tid >> 2)) * KDIM + k0 + 32], 0, 1);
    } else {
      // bf16 source: one async b128 per thread covers the whole 4KB tile
      int m = tid >> 2, part = tid & 3;
      async_ld_b128(&Abuf[m * P + part * 8],
                    Ah + (size_t)(m_blk + m) * KDIM + k0 + part * 8);
    }
    // ---- stage B tile 32x128 transposed -> Bt[n][k] ----
    if constexpr (BF32) {
#pragma unroll
      for (int e = tid; e < 128 * 32; e += 256) {
        int kk = e >> 7, n = e & 127;
        Bt[n * P + kk] = (bf16)Bf[(size_t)(k0 + kk) * NDIM + (n_blk + n)];
      }
      if (k0 + 32 < KDIM)
        __builtin_prefetch(&Bf[(size_t)(k0 + 32 + (tid & 31)) * NDIM + n_blk], 0, 1);
    } else {
#pragma unroll
      for (int e = tid; e < 128 * 32; e += 256) {
        int kk = e >> 7, n = e & 127;
        Bt[n * P + kk] = Bh[(size_t)(k0 + kk) * NDIM + (n_blk + n)];
      }
    }
    if constexpr (!AF32) wait_async0();
    __syncthreads();

    v16bf a0 = load_a_frag(Abuf, P, mw * 32 + (lane & 15), lane);
    v16bf a1 = load_a_frag(Abuf, P, mw * 32 + 16 + (lane & 15), lane);
    v16bf b0 = load_b_frag(Bt, P, nw * 32 + (lane & 15), lane);
    v16bf b1 = load_b_frag(Bt, P, nw * 32 + 16 + (lane & 15), lane);
    acc[0][0] = wmma_bf16(a0, b0, acc[0][0]);
    acc[0][1] = wmma_bf16(a0, b1, acc[0][1]);
    acc[1][0] = wmma_bf16(a1, b0, acc[1][0]);
    acc[1][1] = wmma_bf16(a1, b1, acc[1][1]);
  }

  // Epilogue. C/D: lanes 0-15 -> N=lane, M=v ; lanes 16-31 -> N=lane-16, M=8+v
#pragma unroll
  for (int tm = 0; tm < 2; ++tm) {
#pragma unroll
    for (int tn = 0; tn < 2; ++tn) {
      v8f accT = acc[tm][tn];
      int n = n_blk + nw * 32 + tn * 16 + (lane & 15);
      float bvv = BIAS ? bias[n] : 0.0f;
#pragma unroll
      for (int vi = 0; vi < 8; ++vi) {
        int m = m_blk + mw * 32 + tm * 16 + ((lane & 16) ? (8 + vi) : vi);
        float val = accT[vi] + bvv;
        if (OUTF32)
          ((float*)outp)[(size_t)m * NDIM + n] = val;
        else
          ((bf16*)outp)[(size_t)m * NDIM + n] = (bf16)val;
      }
    }
  }
}

// ---------------------------------------------------------------------------
// Fused scores + mask + softmax. One WG = one (batch, 64-row q-strip).
// Full 64x512 f32 score strip in LDS (CDNA5 320KB/WGP). q_ strip and k_ chunks
// are staged by the Tensor Data Mover with LDS-pitch padding.
// ---------------------------------------------------------------------------
#define SPITCH  516
#define AQPITCH 520
#define BTPITCH 40
#define S_OFF   0
#define AQ_OFF  132096  /* 64*516*4 */
#define BT_OFF  198656  /* + 64*520*2 ; 16B aligned */
#define RED_OFF 208896  /* + 128*40*2 */
#define SMEM_SZ 210944  /* + 512*4 */

__global__ __launch_bounds__(256) void attn_softmax_kernel(
    const bf16* __restrict__ qb, const bf16* __restrict__ kb,
    const float* __restrict__ mask, float* __restrict__ attn_out) {
  extern __shared__ char smem[];
  float* S   = (float*)(smem + S_OFF);   // 64 x 516 score strip
  bf16*  Aq  = (bf16*)(smem + AQ_OFF);   // 64 x 520 q_ strip (bf16)
  bf16*  Bt  = (bf16*)(smem + BT_OFF);   // 128 x 40 k_ chunk (row-major k_ == B^T)
  float* red = (float*)(smem + RED_OFF); // [0..255] row max, [256..511] row sum

  const int b    = blockIdx.x >> 3;
  const int m0   = (blockIdx.x & 7) * 64;  // query offset within batch
  const int tid  = threadIdx.x;
  const int lane = tid & 31;
  const int w    = tid >> 5;
  const int mw   = w & 1;
  const int nw   = w >> 1;
  const float scale = 0.044194173824159216f;  // 1/sqrt(512)

  // TDM: stage the whole 64x512 q_ strip once. Row = 256 DWORDs, pad 4 DWORDs
  // -> pitch 260 DW = 520 bf16 (pad_interval=7 -> 256 DW, pad_amount=3 -> 4 DW).
  // Wait is deferred: TDM ops are in-order per wave, the first chunk wait
  // below covers it.
  if (w == 0) {
    tdm_load_2d((unsigned)(uintptr_t)Aq,
                qb + ((size_t)b * 512 + m0) * 512,
                /*tile*/ 512, 64, /*tensor*/ 512, 512, /*stride*/ 512,
                /*pad_interval*/ 7, /*pad_amount*/ 3);
  }

  // scores(m,n) = sum_d q_[m,d] * k_[n,d]; process 128 key-columns per pass.
  for (int np = 0; np < 4; ++np) {
    const int n0 = np * 128;
    v8f acc[2][2] = {};
    for (int d0 = 0; d0 < 512; d0 += 32) {
      __syncthreads();  // everyone done reading previous Bt
      if (w == 0) {
        // TDM: k_ rows [n0, n0+128) x [d0, d0+32). Row = 16 DWORDs, pad 4
        // -> pitch 20 DW = 40 bf16 (pad_interval=3 -> 16 DW, pad_amount=3).
        tdm_load_2d((unsigned)(uintptr_t)Bt,
                    kb + ((size_t)b * 512 + n0) * 512 + d0,
                    /*tile*/ 32, 128, /*tensor*/ 512, 512, /*stride*/ 512,
                    /*pad_interval*/ 3, /*pad_amount*/ 3);
        wait_tensor0();
      }
      __syncthreads();

      v16bf a0 = load_a_frag(Aq + d0, AQPITCH, mw * 32 + (lane & 15), lane);
      v16bf a1 = load_a_frag(Aq + d0, AQPITCH, mw * 32 + 16 + (lane & 15), lane);
      v16bf b0 = load_b_frag(Bt, BTPITCH, nw * 32 + (lane & 15), lane);
      v16bf b1 = load_b_frag(Bt, BTPITCH, nw * 32 + 16 + (lane & 15), lane);
      acc[0][0] = wmma_bf16(a0, b0, acc[0][0]);
      acc[0][1] = wmma_bf16(a0, b1, acc[0][1]);
      acc[1][0] = wmma_bf16(a1, b0, acc[1][0]);
      acc[1][1] = wmma_bf16(a1, b1, acc[1][1]);
    }
    // scale + structural mask into the LDS score strip
#pragma unroll
    for (int tm = 0; tm < 2; ++tm) {
#pragma unroll
      for (int tn = 0; tn < 2; ++tn) {
        v8f accT = acc[tm][tn];
        int ncol = n0 + nw * 32 + tn * 16 + (lane & 15);
#pragma unroll
        for (int vi = 0; vi < 8; ++vi) {
          int m = mw * 32 + tm * 16 + ((lane & 16) ? (8 + vi) : vi);
          float mk = mask[(size_t)(m0 + m) * 512 + ncol];
          S[m * SPITCH + ncol] = accT[vi] * scale + mk * -1e9f;
        }
      }
    }
  }
  __syncthreads();

  // Row-wise softmax: 4 threads per row, 128 columns each.
  const int r = tid >> 2, part = tid & 3;
  float* srow = S + r * SPITCH + part * 128;
  float pm = -3.402823466e38f;
#pragma unroll 8
  for (int i = 0; i < 128; ++i) pm = fmaxf(pm, srow[i]);
  red[r * 4 + part] = pm;
  __syncthreads();
  float rmax = fmaxf(fmaxf(red[r * 4], red[r * 4 + 1]),
                     fmaxf(red[r * 4 + 2], red[r * 4 + 3]));
  float ps = 0.0f;
#pragma unroll 8
  for (int i = 0; i < 128; ++i) {
    float ev = __expf(srow[i] - rmax);
    srow[i] = ev;
    ps += ev;
  }
  red[256 + r * 4 + part] = ps;
  __syncthreads();
  float inv = 1.0f / (red[256 + r * 4] + red[256 + r * 4 + 1] +
                      red[256 + r * 4 + 2] + red[256 + r * 4 + 3]);
  size_t obase = ((size_t)b * 512 + m0 + r) * 512 + (size_t)part * 128;
#pragma unroll 8
  for (int i = 0; i < 128; ++i) attn_out[obase + i] = srow[i] * inv;
}

// ---------------------------------------------------------------------------
// Launch: 3 projections -> fused scores/softmax -> attn@v_ -> output proj.
// ---------------------------------------------------------------------------
extern "C" void kernel_launch(void* const* d_in, const int* in_sizes, int n_in,
                              void* d_out, int out_size, void* d_ws,
                              size_t ws_size, hipStream_t stream) {
  (void)in_sizes; (void)n_in; (void)out_size; (void)ws_size;
  const float* q    = (const float*)d_in[0];
  const float* k    = (const float*)d_in[1];
  const float* v    = (const float*)d_in[2];
  const float* mask = (const float*)d_in[3];
  const float* wq   = (const float*)d_in[4];
  const float* bq   = (const float*)d_in[5];
  const float* wk   = (const float*)d_in[6];
  const float* bk   = (const float*)d_in[7];
  const float* wv   = (const float*)d_in[8];
  const float* bv   = (const float*)d_in[9];
  const float* wo   = (const float*)d_in[10];
  const float* bo   = (const float*)d_in[11];

  const size_t ELEMS = (size_t)64 * 512 * 512;  // 16,777,216 per [B,N,512]
  bf16* qb   = (bf16*)d_ws;  // bf16 intermediates in workspace
  bf16* kbuf = qb + ELEMS;
  bf16* vbuf = kbuf + ELEMS;
  bf16* ctxb = vbuf + ELEMS;

  float* out_f  = (float*)d_out;  // [B,N,D_MODEL] fp32
  float* attn_f = out_f + ELEMS;  // [B,N,N] fp32 (tuple part 2)

  dim3 grid(512, 4), blk(256);
  // Projections: fp32 A, fp32 W, +bias, bf16 out
  gemm512<true, true, false, true, false><<<grid, blk, 0, stream>>>(q, wq, bq, qb);
  gemm512<true, true, false, true, false><<<grid, blk, 0, stream>>>(k, wk, bk, kbuf);
  gemm512<true, true, false, true, false><<<grid, blk, 0, stream>>>(v, wv, bv, vbuf);

  // Fused QK^T * scale + mask + softmax -> attn (fp32, directly into d_out)
  attn_softmax_kernel<<<dim3(512), blk, SMEM_SZ, stream>>>(qb, kbuf, mask, attn_f);

  // ctx = attn @ v_ : fp32 A (attn), bf16 batched B (v_), no bias, bf16 out
  gemm512<true, false, true, false, false><<<grid, blk, 0, stream>>>(attn_f, vbuf,
                                                                     nullptr, ctxb);
  // out = ctx @ wo + bo : bf16 A (async-staged), fp32 W, +bias, fp32 out
  gemm512<false, true, false, true, true><<<grid, blk, 0, stream>>>(ctxb, wo, bo,
                                                                    out_f);
}
